// Attention_5995774345380
// MI455X (gfx1250) — compile-verified, matrix operand order checked by reading, not run on hardware
//
#include <hip/hip_runtime.h>
#include <hip/hip_bf16.h>

// ---------------------------------------------------------------------------
// Multi-head attention (B=2, N=2048, C=1024, H=16, D=64) on gfx1250 (CDNA5).
// GEMMs on v_wmma_f32_16x16x32_bf16; flash-attention K/V tiles staged with the
// Tensor Data Mover (tensor_load_to_lds + s_wait_tensorcnt).
// ---------------------------------------------------------------------------

typedef __attribute__((ext_vector_type(16))) __bf16 v16bf;
typedef __attribute__((ext_vector_type(8)))  float v8f;
typedef __attribute__((ext_vector_type(4)))  unsigned int v4u;
typedef __attribute__((ext_vector_type(8)))  unsigned int v8u;

union V16 { v16bf v; unsigned short u[16]; };
union V8  { v8f  v; float f[8]; };

static __device__ __forceinline__ unsigned short f2bf(float f) {
    // round-to-nearest-even fp32 -> bf16
    unsigned u = __builtin_bit_cast(unsigned, f);
    u += 0x7FFFu + ((u >> 16) & 1u);
    return (unsigned short)(u >> 16);
}

// ISA 7.12.2: 16-bit A-matrix 16x32 lane/VGPR layout. Element e (0..15) of a
// lane's v16bf operand holds K index:
//   j = e>>1 (VGPR), K = (j>>2)*16 + (j&3)*2 + (laneHi)*8 + (e&1)
// Elements 0..7 / 8..15 form contiguous K-runs -> each fuses to ds_load_b128.
static __device__ __forceinline__ int kmap(int e, int hi) {
    int j = e >> 1;
    return ((j >> 2) << 4) | ((j & 3) << 1) | (hi << 3) | (e & 1);
}

// ---------------------------------------------------------------------------
// TDM: DMA a 64-row x 128-byte tile (64x64 bf16) global -> LDS, with 16B of
// LDS padding after each row (row pitch 144B = 72 bf16). D# per ISA ch.8:
//   group0: count=1 | lds_addr | global_addr | type=2
//   group1: data_size=8B, pad every 32 DWORDs by 4 DWORDs,
//           tensor_dim0=16 units, dim0_stride=16, tensor_dim1=2048 rows,
//           tile_dim0=16 units, tile_dim1=64 rows (2-D tile)
//   group2/3: zero (dims 2..4 unused)
// ---------------------------------------------------------------------------
static __device__ __forceinline__ void tdm_load_tile64x64(
    const unsigned short* gptr, unsigned lds_byte_addr)
{
    const unsigned long long ga = (unsigned long long)gptr;
    v4u g0;
    g0[0] = 1u;                                   // count=1, user descriptor
    g0[1] = lds_byte_addr;                        // lds_addr[31:0]
    g0[2] = (unsigned)ga;                         // global_addr[31:0]
    g0[3] = (unsigned)(ga >> 32) | 0x80000000u;   // global_addr[56:32] | type=2
    v8u g1;
    g1[0] = (3u << 16)    // data_size = 8 bytes
          | (1u << 20)    // pad_enable
          | (4u << 22)    // pad_interval: every 32 DWORDs (one 128B row)
          | (3u << 25);   // pad_amount: 4 DWORDs (16B)
    g1[1] = (16u << 16);    // tensor_dim0 = 16 units (bits 79:48, low part)
    g1[2] = (2048u << 16);  // tensor_dim1 = 2048 rows (bits 111:80, low part)
    g1[3] = (16u << 16);    // tile_dim0 = 16 units (bits 127:112)
    g1[4] = 64u;            // tile_dim1 = 64 rows (bits 143:128); tile_dim2 = 0
    g1[5] = 16u;            // tensor_dim0_stride = 16 units (bits 207:160)
    g1[6] = 0u;             // dim0_stride hi / dim1_stride lo
    g1[7] = 0u;             // dim1_stride hi (unused for 2-D tile)
    const v4u gz = {0u, 0u, 0u, 0u};
    asm volatile("tensor_load_to_lds %0, %1, %2, %3"
                 :: "s"(g0), "s"(g1), "s"(gz), "s"(gz)
                 : "memory");
}

// ===========================================================================
// Generic bf16 WMMA GEMM:  C[M x Ncols] = A[M x K] * B[K x Ncols]
// Block: 256 threads (8 waves). Block tile 128(M) x 128(N); K stepped by 64.
// Wave w owns a 32x64 strip: rows (w&3)*32, cols (w>>2)*64 -> 8 accumulators,
// 16 WMMA per K-stage between barriers.
// MODE 0: QKV epilogue (scatter into q/k/v bf16 [B,H,N,D], q scaled).
// MODE 1: out-proj epilogue (add bias, write fp32).
// ===========================================================================
template <int MODE>
__global__ __launch_bounds__(256) void gemm_bf16_kernel(
    const float* __restrict__ A, const float* __restrict__ Bm,
    int M, int Ncols, int K,
    unsigned short* __restrict__ qb, unsigned short* __restrict__ kb,
    unsigned short* __restrict__ vb,
    const float* __restrict__ bias, float* __restrict__ out)
{
    __shared__ unsigned short As[128 * 72];   // 128 rows x 64 K, pitch 72
    __shared__ unsigned short Bs[64 * 136];   // 64 K x 128 N, pitch 136

    const int tid  = threadIdx.x;
    const int wave = tid >> 5;
    const int lane = tid & 31;
    const int m    = lane & 15;
    const int hi   = lane >> 4;
    const int rw   = wave & 3;       // row group: rows rw*32 .. rw*32+31
    const int cw   = wave >> 2;      // col group: cols cw*64 .. cw*64+63
    const int m0   = blockIdx.y * 128;
    const int n0   = blockIdx.x * 128;

    V8 acc[2][4];
#pragma unroll
    for (int rt = 0; rt < 2; ++rt)
#pragma unroll
        for (int ct = 0; ct < 4; ++ct)
#pragma unroll
            for (int r = 0; r < 8; ++r) acc[rt][ct].f[r] = 0.0f;

    for (int k0 = 0; k0 < K; k0 += 64) {
        // ---- stage A tile (128x64 fp32 -> bf16 LDS): 2048 float4, 8/thread ----
        for (int i = tid; i < 2048; i += 256) {
            int r = i >> 4, c4 = (i & 15) << 2;
            const float4 f = *(const float4*)(A + (size_t)(m0 + r) * K + k0 + c4);
            uint2 p;
            p.x = (unsigned)f2bf(f.x) | ((unsigned)f2bf(f.y) << 16);
            p.y = (unsigned)f2bf(f.z) | ((unsigned)f2bf(f.w) << 16);
            *(uint2*)&As[r * 72 + c4] = p;
        }
        // ---- stage B tile (64x128 fp32 -> bf16 LDS): 2048 float4, 8/thread ----
        for (int i = tid; i < 2048; i += 256) {
            int r = i >> 5, c4 = (i & 31) << 2;
            const float4 f = *(const float4*)(Bm + (size_t)(k0 + r) * Ncols + n0 + c4);
            uint2 p;
            p.x = (unsigned)f2bf(f.x) | ((unsigned)f2bf(f.y) << 16);
            p.y = (unsigned)f2bf(f.z) | ((unsigned)f2bf(f.w) << 16);
            *(uint2*)&Bs[r * 136 + c4] = p;
        }
        __syncthreads();

        if (k0 + 64 < K) {  // hint next A tile into cache
            __builtin_prefetch(A + (size_t)(m0 + (tid >> 1)) * K + k0 + 64, 0, 1);
        }

        // ---- compute: per K-chunk 2 A ops + 4 B ops -> 8 WMMA (16 total) ----
#pragma unroll
        for (int kc = 0; kc < 2; ++kc) {
            V16 a[2];
#pragma unroll
            for (int rt = 0; rt < 2; ++rt)
#pragma unroll
                for (int e = 0; e < 16; ++e)
                    a[rt].u[e] = As[(rw * 32 + rt * 16 + m) * 72 + kc * 32 + kmap(e, hi)];
#pragma unroll
            for (int ct = 0; ct < 4; ++ct) {
                V16 b;
#pragma unroll
                for (int e = 0; e < 16; ++e)
                    b.u[e] = Bs[(kc * 32 + kmap(e, hi)) * 136 + cw * 64 + ct * 16 + m];
#pragma unroll
                for (int rt = 0; rt < 2; ++rt) {
                    acc[rt][ct].v = __builtin_amdgcn_wmma_f32_16x16x32_bf16(
                        false, a[rt].v, false, b.v, (short)0, acc[rt][ct].v, false, false);
                }
            }
        }
        __syncthreads();
    }

    // ---- epilogue ----
#pragma unroll
    for (int rt = 0; rt < 2; ++rt) {
#pragma unroll
        for (int ct = 0; ct < 4; ++ct) {
#pragma unroll
            for (int r = 0; r < 8; ++r) {
                const int row = m0 + rw * 32 + rt * 16 + r + (hi << 3);
                const int col = n0 + cw * 64 + ct * 16 + m;
                const float val = acc[rt][ct].f[r];
                if constexpr (MODE == 0) {
                    // col -> [which(3)][h(16)][d(64)]; row -> [b(2)][n(2048)]
                    const int which = col >> 10;
                    const int rem   = col & 1023;
                    const int h     = rem >> 6;
                    const int d     = rem & 63;
                    const int bb    = row >> 11;
                    const int n     = row & 2047;
                    unsigned short* dst = (which == 0) ? qb : (which == 1) ? kb : vb;
                    const float sv = (which == 0) ? val * 0.125f : val;  // DIM_HEAD^-0.5
                    dst[((((size_t)bb << 4) + h) * 2048 + n) * 64 + d] = f2bf(sv);
                } else {
                    out[(size_t)row * Ncols + col] = val + bias[col];
                }
            }
        }
    }
}

// ===========================================================================
// Flash attention: one block = one (b,h) x 64 q-rows. 128 threads (4 waves),
// wave w owns q rows [w*16, w*16+16). kv swept in 64-wide tiles with online
// softmax; K/V tiles DMA'd global->LDS by the Tensor Data Mover.
// S = Q*K^T and O += P*V both on bf16 WMMA (16 WMMA per kv tile).
// ===========================================================================
__global__ __launch_bounds__(128) void flash_attn_kernel(
    const unsigned short* __restrict__ Qw, const unsigned short* __restrict__ Kw,
    const unsigned short* __restrict__ Vw, float* __restrict__ O)
{
    __shared__ unsigned short Qs[64 * 72];      // 64 q rows x 64 d
    __shared__ unsigned short Ks[64 * 72];      // 64 kv rows x 64 d (TDM dest)
    __shared__ unsigned short Vs[64 * 72];      // 64 kv rows x 64 d (TDM dest)
    __shared__ unsigned short Ps[4 * 16 * 72];  // per-wave 16 x 64 P tile

    const int tid  = threadIdx.x;
    const int wave = tid >> 5;
    const int lane = tid & 31;
    const int m    = lane & 15;
    const int hi   = lane >> 4;
    const int bh   = blockIdx.y;        // b*16 + h
    const int q0   = blockIdx.x * 64;

    // addrspace(3) byte offsets of the TDM destination tiles
    const unsigned ksLds = (unsigned)(unsigned long long)(const void*)&Ks[0];
    const unsigned vsLds = (unsigned)(unsigned long long)(const void*)&Vs[0];

    // ---- stage Q tile (bf16, 8B vector copies) ----
    {
        const uint2* src = (const uint2*)(Qw + ((size_t)bh * 2048 + q0) * 64);
        for (int i = tid; i < 1024; i += 128) {
            int r = i >> 4, c4 = (i & 15) << 2;
            *(uint2*)&Qs[r * 72 + c4] = src[i];
        }
    }

    V8 Oacc[4];
    float mi[8], li[8];
#pragma unroll
    for (int t = 0; t < 4; ++t)
#pragma unroll
        for (int r = 0; r < 8; ++r) Oacc[t].f[r] = 0.0f;
#pragma unroll
    for (int r = 0; r < 8; ++r) { mi[r] = -1e30f; li[r] = 0.0f; }

    for (int kv0 = 0; kv0 < 2048; kv0 += 64) {
        // ---- TDM: DMA K and V 64x64 tiles into padded LDS (wave 0 issues) ----
        if (wave == 0) {
            tdm_load_tile64x64(Kw + ((size_t)bh * 2048 + kv0) * 64, ksLds);
            tdm_load_tile64x64(Vw + ((size_t)bh * 2048 + kv0) * 64, vsLds);
            asm volatile("s_wait_tensorcnt 0x0" ::: "memory");
        }
        __syncthreads();

        // ---- S[16 x 64] = Q_strip * K^T : 8 WMMA (2 K-chunks x 4 kv tiles) ----
        V8 S[4];
#pragma unroll
        for (int t = 0; t < 4; ++t)
#pragma unroll
            for (int r = 0; r < 8; ++r) S[t].f[r] = 0.0f;

#pragma unroll
        for (int dc = 0; dc < 2; ++dc) {
            V16 aq;
#pragma unroll
            for (int e = 0; e < 16; ++e)
                aq.u[e] = Qs[(wave * 16 + m) * 72 + dc * 32 + kmap(e, hi)];
#pragma unroll
            for (int t = 0; t < 4; ++t) {
                V16 bk;  // B = K^T: lane n holds kv row (t*16+n) along d
#pragma unroll
                for (int e = 0; e < 16; ++e)
                    bk.u[e] = Ks[(t * 16 + m) * 72 + dc * 32 + kmap(e, hi)];
                S[t].v = __builtin_amdgcn_wmma_f32_16x16x32_bf16(
                    false, aq.v, false, bk.v, (short)0, S[t].v, false, false);
            }
        }

        // ---- online softmax (rows live in VGPRs, cols in 16-lane groups) ----
        float alpha[8];
#pragma unroll
        for (int r = 0; r < 8; ++r) {
            float v = S[0].f[r];
#pragma unroll
            for (int t = 1; t < 4; ++t) v = fmaxf(v, S[t].f[r]);
            v = fmaxf(v, __shfl_xor(v, 1, 16));
            v = fmaxf(v, __shfl_xor(v, 2, 16));
            v = fmaxf(v, __shfl_xor(v, 4, 16));
            v = fmaxf(v, __shfl_xor(v, 8, 16));
            const float mnew = fmaxf(mi[r], v);
            alpha[r] = __expf(mi[r] - mnew);
            mi[r] = mnew;
        }
#pragma unroll
        for (int r = 0; r < 8; ++r) {
            float s = 0.0f;
#pragma unroll
            for (int t = 0; t < 4; ++t) {
                S[t].f[r] = __expf(S[t].f[r] - mi[r]);
                s += S[t].f[r];
            }
            s += __shfl_xor(s, 1, 16);
            s += __shfl_xor(s, 2, 16);
            s += __shfl_xor(s, 4, 16);
            s += __shfl_xor(s, 8, 16);
            li[r] = li[r] * alpha[r] + s;
        }
#pragma unroll
        for (int t = 0; t < 4; ++t)
#pragma unroll
            for (int r = 0; r < 8; ++r) Oacc[t].f[r] *= alpha[r];

        // ---- D-layout -> A-layout transpose of P through LDS ----
#pragma unroll
        for (int t = 0; t < 4; ++t)
#pragma unroll
            for (int r = 0; r < 8; ++r)
                Ps[wave * 1152 + (r + (hi << 3)) * 72 + t * 16 + m] = f2bf(S[t].f[r]);
        __syncthreads();

        // ---- O[16 x 64] += P[16 x 64] * V[64 x 64] : 8 WMMA ----
        V16 ap0, ap1;
#pragma unroll
        for (int e = 0; e < 16; ++e) {
            ap0.u[e] = Ps[wave * 1152 + m * 72 +      kmap(e, hi)];
            ap1.u[e] = Ps[wave * 1152 + m * 72 + 32 + kmap(e, hi)];
        }
#pragma unroll
        for (int dt = 0; dt < 4; ++dt) {
            V16 bv0, bv1;  // B = V: lane n holds d column (dt*16+n) along kv
#pragma unroll
            for (int e = 0; e < 16; ++e) {
                bv0.u[e] = Vs[(     kmap(e, hi)) * 72 + dt * 16 + m];
                bv1.u[e] = Vs[(32 + kmap(e, hi)) * 72 + dt * 16 + m];
            }
            Oacc[dt].v = __builtin_amdgcn_wmma_f32_16x16x32_bf16(
                false, ap0.v, false, bv0.v, (short)0, Oacc[dt].v, false, false);
            Oacc[dt].v = __builtin_amdgcn_wmma_f32_16x16x32_bf16(
                false, ap1.v, false, bv1.v, (short)0, Oacc[dt].v, false, false);
        }
        __syncthreads();
    }

    // ---- finalize: O /= l, write fp32 [B, N, H*D] ----
    const int b = bh >> 4;
    const int h = bh & 15;
#pragma unroll
    for (int dt = 0; dt < 4; ++dt) {
#pragma unroll
        for (int r = 0; r < 8; ++r) {
            const int n   = q0 + wave * 16 + r + (hi << 3);
            const int col = (h << 6) + dt * 16 + m;
            O[((size_t)b * 2048 + n) * 1024 + col] = Oacc[dt].f[r] / li[r];
        }
    }
}

// ===========================================================================
extern "C" void kernel_launch(void* const* d_in, const int* in_sizes, int n_in,
                              void* d_out, int out_size, void* d_ws, size_t ws_size,
                              hipStream_t stream) {
    const float* x      = (const float*)d_in[0];  // [2, 2048, 1024]
    const float* w_qkv  = (const float*)d_in[1];  // [1024, 3072]
    const float* w_proj = (const float*)d_in[2];  // [1024, 1024]
    const float* b_proj = (const float*)d_in[3];  // [1024]
    float* out = (float*)d_out;                   // [2, 2048, 1024] fp32

    // workspace: q/k/v bf16 [B,H,N,D] (8 MB each) + attn out fp32 (16 MB)
    char* ws = (char*)d_ws;
    const size_t QKV_ELEMS = (size_t)2 * 16 * 2048 * 64;  // 4 Mi elements
    unsigned short* qb = (unsigned short*)ws;
    unsigned short* kb = qb + QKV_ELEMS;
    unsigned short* vb = kb + QKV_ELEMS;
    float* attn = (float*)(ws + 3 * QKV_ELEMS * sizeof(unsigned short));

    // 1) QKV projection + head split (M=4096, N=3072, K=1024)
    gemm_bf16_kernel<0><<<dim3(24, 32), 256, 0, stream>>>(
        x, w_qkv, 4096, 3072, 1024, qb, kb, vb, nullptr, nullptr);

    // 2) flash attention: 32 q-tiles x 32 (b,h)
    flash_attn_kernel<<<dim3(32, 32), 128, 0, stream>>>(qb, kb, vb, attn);

    // 3) output projection + bias (M=4096, N=1024, K=1024)
    gemm_bf16_kernel<1><<<dim3(8, 32), 256, 0, stream>>>(
        attn, w_proj, 4096, 1024, 1024, nullptr, nullptr, nullptr, b_proj, out);
}
